// EmbeddingHead_80204219285824
// MI455X (gfx1250) — matrix operand order repaired and checked by reading, not compiled, not run on hardware
//
#include <hip/hip_runtime.h>
#include <hip/hip_bf16.h>

// ---------------------------------------------------------------------------
// EmbeddingHead for MI455X (gfx1250, wave32, WMMA)
//   out = [logits(256x10000), logits(256x10000), pooled(256x2048), ones(256)]
// Softmax/target chain is numerically dead (tanh(+inf)=1 -> new_weight==1.0).
// GEMM: bf16 WMMA (v_wmma_f32_16x16x32_bf16) with f32 accumulation.
// Each wave: 2 N-tiles x 4 M-tiles = 8 WMMAs per 32-K step.
// ---------------------------------------------------------------------------

typedef __attribute__((ext_vector_type(16))) __bf16 v16bf;
typedef __attribute__((ext_vector_type(8)))  float  v8f;

#define C_DIM 2048
#define HW    128
#define B_DIM 256
#define N_CLS 10000
#define BM    64      // block M tile (4 x 16 per wave)
#define KT    32      // K per WMMA step
#define BN    256     // block N tile (8 waves x 2 x 16)

union Frag16 { v16bf v; uint4 q[2]; unsigned int u[8]; };

// f32 pair -> packed bf16 dword (lo=a, hi=b)
#if __has_builtin(__builtin_amdgcn_cvt_pk_bf16_f32)
__device__ __forceinline__ unsigned int pack2_bf16(float a, float b) {
  auto t = __builtin_amdgcn_cvt_pk_bf16_f32(a, b);   // v_cvt_pk_bf16_f32 (RNE)
  unsigned int u;
  __builtin_memcpy(&u, &t, 4);
  return u;
}
#else
__device__ __forceinline__ unsigned int pack2_bf16(float a, float b) {
  const unsigned int ua = __float_as_uint(a) + 0x8000u;  // round-half-up
  const unsigned int ub = __float_as_uint(b) + 0x8000u;
  return __builtin_amdgcn_perm(ub, ua, 0x07060302u);     // {ub.hi16, ua.hi16}
}
#endif

__device__ __forceinline__ v8f wmma_bf16(const Frag16& a, const Frag16& b, v8f c) {
  return __builtin_amdgcn_wmma_f32_16x16x32_bf16(
      false, a.v, false, b.v, (short)0, c, false, false);
}

// ---------------- global average pool: one wave per (b,c) row --------------
__global__ void pool_kernel(const float* __restrict__ feat,
                            float* __restrict__ Aout,
                            float* __restrict__ featOut) {
  const int gid  = blockIdx.x * blockDim.x + threadIdx.x;
  const int row  = gid >> 5;        // (b*C + c), 524288 rows
  const int lane = gid & 31;
  const float4 v = ((const float4*)(feat + (size_t)row * HW))[lane]; // 512B/wave
  float s = (v.x + v.y) + (v.z + v.w);
#pragma unroll
  for (int off = 16; off > 0; off >>= 1) s += __shfl_xor(s, off, 32);
  if (lane == 0) {
    const float m = s * (1.0f / (float)HW);
    Aout[row]    = m;   // workspace: GEMM A operand
    featOut[row] = m;   // output #3
  }
}

// ---------------- bf16 WMMA GEMM: logits = A[256,2048] x W[10000,2048]^T ---
__global__ void __launch_bounds__(256, 2)
gemm_wmma_kernel(const float* __restrict__ A,
                 const float* __restrict__ Wt,
                 float* __restrict__ out0,
                 float* __restrict__ out1) {
  __shared__ unsigned int Asm[2][BM * (KT / 2)];   // double-buffered bf16 A tile

  const int tid  = threadIdx.x;
  const int wave = tid >> 5;
  const int lane = tid & 31;
  const int m0   = blockIdx.y * BM;
  const int n0a  = blockIdx.x * BN + wave * 16;    // N tile A
  const int n0b  = n0a + 128;                      // N tile B
  const bool okA = (n0a < N_CLS);
  const bool okB = (n0b < N_CLS);

  // cooperative A staging: 64 rows x 32 K fp32 -> bf16 LDS, 8 floats/thread
  const int lr   = tid >> 2;                       // tile row 0..63
  const int lk   = (tid & 3) * 8;                  // K offset 0/8/16/24
  const float* aptr = A + (size_t)(m0 + lr) * C_DIM + lk;
  const int sidx = lr * (KT / 2) + (tid & 3) * 4;  // dword index, 16B aligned

  // B operand: lane -> column n, two 8-K runs at kbB and kbB+16
  const int kbB = (lane >> 4) * 8;
  const float* wptrA = Wt + (size_t)(okA ? (n0a + (lane & 15)) : 0) * C_DIM + kbB;
  const float* wptrB = Wt + (size_t)(okB ? (n0b + (lane & 15)) : 0) * C_DIM + kbB;

  // A fragment geometry: lane -> row m, two 8-K runs at kbA and kbA+16
  const int mA  = lane & 15;
  const int kbA = (lane >> 4) * 8;

  v8f accA[4] = {};
  v8f accB[4] = {};

  // stage one 64x32 fp32 -> bf16 tile into LDS buffer `buf`
  auto stage = [&](int kk, int buf) {
    const float4 f0 = *(const float4*)(aptr + kk + 0);
    const float4 f1 = *(const float4*)(aptr + kk + 4);
    uint4 p;
    p.x = pack2_bf16(f0.x, f0.y);  p.y = pack2_bf16(f0.z, f0.w);
    p.z = pack2_bf16(f1.x, f1.y);  p.w = pack2_bf16(f1.z, f1.w);
    *(uint4*)&Asm[buf][sidx] = p;
  };

  // load + convert one B fragment (K = kbB..kbB+7, kbB+16..kbB+23 fp32)
  auto loadB = [&](const float* wp, int kk) {
    Frag16 bf;
    const float4 g0 = *(const float4*)(wp + kk + 0);
    const float4 g1 = *(const float4*)(wp + kk + 4);
    const float4 g2 = *(const float4*)(wp + kk + 16);
    const float4 g3 = *(const float4*)(wp + kk + 20);
    bf.u[0] = pack2_bf16(g0.x, g0.y);  bf.u[1] = pack2_bf16(g0.z, g0.w);
    bf.u[2] = pack2_bf16(g1.x, g1.y);  bf.u[3] = pack2_bf16(g1.z, g1.w);
    bf.u[4] = pack2_bf16(g2.x, g2.y);  bf.u[5] = pack2_bf16(g2.z, g2.w);
    bf.u[6] = pack2_bf16(g3.x, g3.y);  bf.u[7] = pack2_bf16(g3.z, g3.w);
    return bf;
  };

  // one 32-K compute step from LDS buffer `buf`
  auto compute = [&](int kk, int buf) {
    __builtin_prefetch(wptrA + kk + 2 * KT, 0, 1);   // global_prefetch_b8
    __builtin_prefetch(wptrB + kk + 2 * KT, 0, 1);
    const Frag16 bA = loadB(wptrA, kk);
    const Frag16 bB = loadB(wptrB, kk);
    Frag16 af[4];
#pragma unroll
    for (int mt = 0; mt < 4; ++mt) {                 // 8 ds_load_b128, 1 wait
      const int rb = (mt * 16 + mA) * (KT / 2) + (kbA >> 1);
      af[mt].q[0] = *(const uint4*)&Asm[buf][rb];      // K = kbA .. kbA+7
      af[mt].q[1] = *(const uint4*)&Asm[buf][rb + 8];  // K = kbA+16 .. kbA+23
    }
#pragma unroll
    for (int mt = 0; mt < 4; ++mt) {                 // 8 back-to-back WMMAs
      accA[mt] = wmma_bf16(af[mt], bA, accA[mt]);
      accB[mt] = wmma_bf16(af[mt], bB, accB[mt]);
    }
  };

  stage(0, 0);
  __syncthreads();

  for (int kk = 0; kk < C_DIM; kk += 2 * KT) {       // 32 iters, static buf ids
    if (kk + KT < C_DIM)      stage(kk + KT, 1);
    compute(kk, 0);
    __syncthreads();
    if (kk + 2 * KT < C_DIM)  stage(kk + 2 * KT, 0);
    compute(kk + KT, 1);
    __syncthreads();
  }

  // epilogue: C/D layout -> lane(0..15)=N, lane>=16 adds 8 to M; v[r] = M row
  auto store_tiles = [&](int nbase, bool ok, v8f* acc) {
    if (!ok) return;
    const int n      = nbase + (lane & 15);
    const int rowadd = (lane >> 4) * 8;
#pragma unroll
    for (int mt = 0; mt < 4; ++mt) {
#pragma unroll
      for (int r = 0; r < 8; ++r) {
        const size_t o = (size_t)(m0 + mt * 16 + rowadd + r) * N_CLS + n;
        const float val = acc[mt][r];
        out0[o] = val;   // cls_outputs
        out1[o] = val;   // pred_class_logits (SCALE == 1)
      }
    }
  };
  store_tiles(n0a, okA, accA);
  store_tiles(n0b, okB, accB);
}

// new_weight == 1.0 exactly (tanh(mean/0)=tanh(+inf)=1, t*1/sum(1)=1)
__global__ void ones_kernel(float* __restrict__ nw) { nw[threadIdx.x] = 1.0f; }

extern "C" void kernel_launch(void* const* d_in, const int* in_sizes, int n_in,
                              void* d_out, int out_size, void* d_ws, size_t ws_size,
                              hipStream_t stream) {
  const float* features = (const float*)d_in[0];
  // d_in[1] (targets) is numerically dead: new_weight is constant 1.0
  const float* weight   = (const float*)d_in[2];
  float* out   = (float*)d_out;
  float* Apool = (float*)d_ws;                        // 2 MB pooled features

  float* cls     = out;                               // [256][10000]
  float* pred    = cls + (size_t)B_DIM * N_CLS;       // [256][10000]
  float* featOut = pred + (size_t)B_DIM * N_CLS;      // [256][2048]
  float* nw      = featOut + (size_t)B_DIM * C_DIM;   // [256]

  pool_kernel<<<(B_DIM * C_DIM) / 8, 256, 0, stream>>>(features, Apool, featOut);

  dim3 grid((N_CLS + BN - 1) / BN, B_DIM / BM);       // 40 x 4
  gemm_wmma_kernel<<<grid, 256, 0, stream>>>(Apool, weight, cls, pred);

  ones_kernel<<<1, B_DIM, 0, stream>>>(nw);
}
